// FlashAttentionWithRoPE_9947144258210
// MI455X (gfx1250) — matrix operand order repaired and checked by reading, compiled, not run on hardware
//
#include <hip/hip_runtime.h>
#include <hip/hip_bf16.h>

// ---------------------------------------------------------------------------
// Types for CDNA5 WMMA (wave32): 16x16x32 bf16 -> f32
// ---------------------------------------------------------------------------
typedef __attribute__((ext_vector_type(16))) __bf16 v16bf;
typedef __attribute__((ext_vector_type(8)))  __bf16 v8bf;
typedef __attribute__((ext_vector_type(8)))  float  v8f;

static __device__ __forceinline__ v8f wmma_bf16(v16bf a, v16bf b, v8f c) {
    // (neg_a, A, neg_b, B, c_mod, C, reuse_a, reuse_b)
    return __builtin_amdgcn_wmma_f32_16x16x32_bf16(false, a, false, b, (short)0, c,
                                                   false, false);
}

// Manual fp32<->bf16 (RNE-ish), avoids any dependence on __hip_bfloat16 ops.
static __device__ __forceinline__ __bf16 f2bf(float f) {
    union { float f; unsigned u; } x; x.f = f;
    unsigned u = x.u;
    unsigned rb = ((u >> 16) & 1u) + 0x7fffu;
    union { unsigned short s; __bf16 b; } y;
    y.s = (unsigned short)((u + rb) >> 16);
    return y.b;
}
static __device__ __forceinline__ float bf2f(__bf16 b) {
    union { unsigned short s; __bf16 b; } y; y.b = b;
    union { unsigned u; float f; } x; x.u = ((unsigned)y.s) << 16;
    return x.f;
}

// ---------------------------------------------------------------------------
// Fragment loaders (contiguous k order k=16h+e, same for A and B, wave32)
// A 16x32 bf16: lane&15 = M row; half-wave selects K range {0..15,16..31};
//   element e -> K=2*kb+e. Row memory is K-contiguous.
// B 32x16 bf16: lane&15 = N col; half-wave selects K range {0..15,16..31};
//   element e -> K=kr+e. With K-contiguous memory this is one 32B load.
// ---------------------------------------------------------------------------
static __device__ __forceinline__ v16bf load_afrag(const __bf16* rowp, int kb8) {
    v8bf lo = *(const v8bf*)(rowp + 2 * kb8);
    v8bf hi = *(const v8bf*)(rowp + 2 * kb8 + 8);
    v16bf a;
#pragma unroll
    for (int e = 0; e < 8; ++e) { a[e] = lo[e]; a[e + 8] = hi[e]; }
    return a;
}

// ---------------------------------------------------------------------------
// fp32 -> bf16 conversion
// ---------------------------------------------------------------------------
__global__ __launch_bounds__(256) void f32_to_bf16_kernel(const float* __restrict__ in,
                                                          __bf16* __restrict__ out, int n) {
    int i = blockIdx.x * 256 + threadIdx.x;
    if (i < n) out[i] = f2bf(in[i]);
}

// ---------------------------------------------------------------------------
// GEMM: C[M,N] = A[M,K] * W[N,K]^T, bf16 in, f32 accumulate.
// Wave tile 32x32 (4 accumulators), block = 8 waves -> 128(M) x 64(N).
// Both A and W are K-contiguous -> vector loads match WMMA fragments exactly.
// ---------------------------------------------------------------------------
template <bool STORE_BF16>
__global__ __launch_bounds__(256) void gemm_wmma_kernel(const __bf16* __restrict__ A,
                                                        const __bf16* __restrict__ W,
                                                        void* __restrict__ Cp,
                                                        int M, int N, int K) {
    const int lane  = threadIdx.x & 31;
    const int wave  = threadIdx.x >> 5;
    const int wm    = wave & 3;
    const int wn    = wave >> 2;
    const int m0    = blockIdx.y * 128 + wm * 32;
    const int n0    = blockIdx.x * 64 + wn * 32;
    const int row   = lane & 15;
    const int kb8   = (lane >> 4) << 3;   // A-frag K base (0 / 8)
    const int kr16  = (lane >> 4) << 4;   // B-frag K base (0 / 16)
    const int colN  = lane & 15;
    const int rbase = (lane >> 4) << 3;   // C row base (0 / 8)

    v8f acc00 = {}, acc01 = {}, acc10 = {}, acc11 = {};

    const __bf16* A0 = A + (size_t)(m0 + row) * K;
    const __bf16* A1 = A + (size_t)(m0 + 16 + row) * K;
    const __bf16* W0 = W + (size_t)(n0 + colN) * K;
    const __bf16* W1 = W + (size_t)(n0 + 16 + colN) * K;

    for (int k0 = 0; k0 < K; k0 += 32) {
        if (k0 + 256 < K) {   // pull next cachelines toward WGP (global_prefetch_b8)
            __builtin_prefetch(A0 + k0 + 256, 0, 0);
            __builtin_prefetch(W0 + k0 + 256, 0, 0);
        }
        v16bf a0 = load_afrag(A0 + k0, kb8);
        v16bf a1 = load_afrag(A1 + k0, kb8);
        v16bf b0 = *(const v16bf*)(W0 + k0 + kr16);
        v16bf b1 = *(const v16bf*)(W1 + k0 + kr16);
        acc00 = wmma_bf16(a0, b0, acc00);
        acc01 = wmma_bf16(a0, b1, acc01);
        acc10 = wmma_bf16(a1, b0, acc10);
        acc11 = wmma_bf16(a1, b1, acc11);
    }

    // C layout: element v -> (M = rbase+v + 16*mt, N = colN + 16*nt)
#pragma unroll
    for (int mt = 0; mt < 2; ++mt) {
#pragma unroll
        for (int nt = 0; nt < 2; ++nt) {
            const v8f* accp = mt == 0 ? (nt == 0 ? &acc00 : &acc01)
                                      : (nt == 0 ? &acc10 : &acc11);
#pragma unroll
            for (int v = 0; v < 8; ++v) {
                size_t idx = (size_t)(m0 + mt * 16 + rbase + v) * N + (n0 + nt * 16 + colN);
                if (STORE_BF16) ((__bf16*)Cp)[idx] = f2bf((*accp)[v]);
                else            ((float*)Cp)[idx] = (*accp)[v];
            }
        }
    }
}

// ---------------------------------------------------------------------------
// RoPE + scatter: qkv bf16 [B,L,3,H,dh] -> q/k bf16 [B,H,L,dh].
// 1/sqrt(dh) attention scale folded into q. (V handled by v_transpose_kernel.)
// ---------------------------------------------------------------------------
__global__ __launch_bounds__(256) void rope_scatter_kernel(const __bf16* __restrict__ qkvb,
                                                           __bf16* __restrict__ Qb,
                                                           __bf16* __restrict__ Kb) {
    constexpr int L = 2048, H = 16, DH = 128, D = 2048;
    int idx = blockIdx.x * 256 + threadIdx.x;     // total = B*L*H*64 = 2^22
    int i = idx & 63;                              // rotation pair index 0..63
    int h = (idx >> 6) & (H - 1);
    int l = (idx >> 10) & (L - 1);
    int b = idx >> 21;

    const __bf16* base = qkvb + ((size_t)(b * L + l)) * (3 * D) + h * DH;
    float q1 = bf2f(base[i]),     q2 = bf2f(base[i + 64]);
    float k1 = bf2f(base[D + i]), k2 = bf2f(base[D + i + 64]);

    // inv_freq = 10000^(-i/64)  (arange(0,128,2)/128 = i/64)
    float f = (float)l * __expf(-(float)i * (9.210340371976184f / 64.0f));
    float s, c;
    __sincosf(f, &s, &c);
    const float scale = 0.08838834764831845f;      // 1/sqrt(128)

    size_t o = ((size_t)(b * H + h) * L + l) * DH;
    Qb[o + i]      = f2bf((q1 * c - q2 * s) * scale);
    Qb[o + i + 64] = f2bf((q1 * s + q2 * c) * scale);
    Kb[o + i]      = f2bf(k1 * c - k2 * s);
    Kb[o + i + 64] = f2bf(k1 * s + k2 * c);
}

// ---------------------------------------------------------------------------
// V transpose: qkv bf16 [B,L,3,H,dh] (v slice) -> Vt bf16 [B*H, dh, L].
// Key-major V lets attention load B-fragments as single 32B vector loads.
// Classic 32x32 LDS tile, coalesced on both sides.
// ---------------------------------------------------------------------------
__global__ __launch_bounds__(256) void v_transpose_kernel(const __bf16* __restrict__ qkvb,
                                                          __bf16* __restrict__ Vt) {
    constexpr int L = 2048, H = 16, DH = 128, D = 2048;
    __shared__ __bf16 tile[32][33];
    const int bh = blockIdx.z;             // 0..B*H-1
    const int b  = bh / H, h = bh % H;
    const int l0 = blockIdx.y * 32;
    const int d0 = blockIdx.x * 32;
    const int tx = threadIdx.x & 31;
    const int ty = threadIdx.x >> 5;       // 0..7

    const __bf16* src = qkvb + (size_t)(b * L) * (3 * D) + 2 * D + h * DH;
#pragma unroll
    for (int i = 0; i < 4; ++i) {
        int l = l0 + ty + i * 8;
        tile[ty + i * 8][tx] = src[(size_t)l * (3 * D) + d0 + tx];
    }
    __syncthreads();
    __bf16* dst = Vt + (size_t)bh * DH * L;
#pragma unroll
    for (int i = 0; i < 4; ++i) {
        int d = d0 + ty + i * 8;
        dst[(size_t)d * L + l0 + tx] = tile[tx][ty + i * 8];
    }
}

// ---------------------------------------------------------------------------
// Causal flash attention. q/k: bf16 [B,H,L,128]; Vt: bf16 [B*H,128,L].
// Out: bf16 [B,L,D]. Block = 4 waves; each wave owns 16 q-rows.
// 64 keys per step (4 score tiles, 2 P fragments). q0 is forced into an SGPR
// via readfirstlane so all causal-tile conditions are scalar branches: EXEC
// stays all-ones around every WMMA (ISA 7.12 requirement) and no cndmask
// scaffolding is generated for dead subtiles.
// P row stride padded to 72 halves (bank-conflict-free A-frag reloads).
// ---------------------------------------------------------------------------
__global__ __launch_bounds__(128) void flash_attn_kernel(const __bf16* __restrict__ Qb,
                                                         const __bf16* __restrict__ Kb,
                                                         const __bf16* __restrict__ Vt,
                                                         __bf16* __restrict__ Ob) {
    constexpr int L = 2048, H = 16, DH = 128, D = 2048;
    constexpr int PSTR = 72;                       // P LDS row stride (halves)
    __shared__ __bf16 plds[4][16 * PSTR];          // per-wave P tile (C->A transpose)

    const int lane  = threadIdx.x & 31;
    const int wave  = threadIdx.x >> 5;
    const int nq64  = L / 64;                      // 32 q-blocks per (b,h)
    const int bid   = blockIdx.x;
    const int q64   = bid % nq64;
    const int h     = (bid / nq64) % H;
    const int b     = bid / (nq64 * H);
    // Wave-uniform by construction; readfirstlane makes that visible to the
    // compiler -> scalar (s_cmp/s_cbranch) control flow, EXEC untouched.
    const int q0    = __builtin_amdgcn_readfirstlane(q64 * 64 + wave * 16);
    const size_t bh = (size_t)(b * H + h) * L;

    const int row   = lane & 15;
    const int kb8   = (lane >> 4) << 3;
    const int kr16  = (lane >> 4) << 4;
    const int colN  = lane & 15;
    const int rbase = (lane >> 4) << 3;

    // Q fragments: 4 chunks of K=32 over dh=128
    v16bf qf[4];
    const __bf16* Qrow = Qb + (bh + q0 + row) * DH;
#pragma unroll
    for (int kk = 0; kk < 4; ++kk) qf[kk] = load_afrag(Qrow + kk * 32, kb8);

    v8f Oacc[8];
#pragma unroll
    for (int nt = 0; nt < 8; ++nt) Oacc[nt] = (v8f){};
    float mrow[8], lrow[8];
#pragma unroll
    for (int v = 0; v < 8; ++v) { mrow[v] = -1e30f; lrow[v] = 0.0f; }

    // Vt column base for this lane: dh col = nt*16 + colN, key index contiguous
    const __bf16* Vcol = Vt + ((size_t)(b * H + h) * DH + colN) * L;
    __bf16* pw = plds[wave];

    const int qlast = q0 + 15;                     // last (largest) q row of wave
    const int jmax  = qlast >> 6;                  // key tiles of 64 (causal bound)
    for (int j = 0; j <= jmax; ++j) {
        const int kbase = j * 64;

        // ---- S = Q * K^T for up to 64 keys (4 16x16 score tiles) ----
        v8f S[4];
#pragma unroll
        for (int t = 0; t < 4; ++t) {
            if (kbase + t * 16 <= qlast) {         // scalar: subtile intersects causal region
                v8f s = {};
                const __bf16* Kt = Kb + (bh + kbase + t * 16 + colN) * DH;
#pragma unroll
                for (int kk = 0; kk < 4; ++kk) {
                    v16bf bk = *(const v16bf*)(Kt + kk * 32 + kr16);
                    s = wmma_bf16(qf[kk], bk, s);
                }
                if (kbase + t * 16 + 15 > q0) {    // scalar: boundary tile needs mask
#pragma unroll
                    for (int v = 0; v < 8; ++v) {  // per-lane mask (v_cmp/v_cndmask)
                        if (kbase + t * 16 + colN > q0 + rbase + v) s[v] = -1e30f;
                    }
                }
                S[t] = s;
            } else {                               // fully masked (scalar-skipped)
#pragma unroll
                for (int v = 0; v < 8; ++v) S[t][v] = -1e30f;
            }
        }

        // ---- online softmax update over 64 keys ----
        float alpha[8];
#pragma unroll
        for (int v = 0; v < 8; ++v) {
            float mx = fmaxf(fmaxf(S[0][v], S[1][v]), fmaxf(S[2][v], S[3][v]));
#pragma unroll
            for (int msk = 1; msk < 16; msk <<= 1) mx = fmaxf(mx, __shfl_xor(mx, msk, 32));
            float mnew = fmaxf(mrow[v], mx);
            alpha[v] = __expf(mrow[v] - mnew);
            mrow[v] = mnew;
            float p0 = __expf(S[0][v] - mnew);
            float p1 = __expf(S[1][v] - mnew);
            float p2 = __expf(S[2][v] - mnew);
            float p3 = __expf(S[3][v] - mnew);
            S[0][v] = p0; S[1][v] = p1; S[2][v] = p2; S[3][v] = p3;
            float rs = (p0 + p1) + (p2 + p3);
#pragma unroll
            for (int msk = 1; msk < 16; msk <<= 1) rs += __shfl_xor(rs, msk, 32);
            lrow[v] = lrow[v] * alpha[v] + rs;
        }

        // ---- P: C-layout -> A-layout via per-wave LDS tile ----
#pragma unroll
        for (int t = 0; t < 4; ++t) {
#pragma unroll
            for (int v = 0; v < 8; ++v) {
                pw[(rbase + v) * PSTR + t * 16 + colN] = f2bf(S[t][v]);
            }
        }
        asm volatile("s_wait_dscnt 0" ::: "memory");   // wave-internal LDS RAW fence
        v16bf pf0 = load_afrag(pw + row * PSTR, kb8);        // keys kbase+0..31
        v16bf pf1 = load_afrag(pw + row * PSTR + 32, kb8);   // keys kbase+32..63

        // ---- O = O*alpha + P @ V ----
#pragma unroll
        for (int nt = 0; nt < 8; ++nt) {
#pragma unroll
            for (int v = 0; v < 8; ++v) Oacc[nt][v] *= alpha[v];
        }
        // V B-fragment: b[e] = V[key = kbase+koff+kr16+e][dh = nt*16+colN]
        //             = Vt[(nt*16+colN)*L + kbase+koff+kr16+e]  -> one 32B load
#pragma unroll
        for (int nt = 0; nt < 8; ++nt) {
            v16bf vf = *(const v16bf*)(Vcol + (size_t)nt * 16 * L + kbase + kr16);
            Oacc[nt] = wmma_bf16(pf0, vf, Oacc[nt]);
        }
        if (kbase + 32 <= qlast) {                 // scalar: second half has live keys
#pragma unroll
            for (int nt = 0; nt < 8; ++nt) {
                v16bf vf = *(const v16bf*)(Vcol + (size_t)nt * 16 * L + kbase + 32 + kr16);
                Oacc[nt] = wmma_bf16(pf1, vf, Oacc[nt]);
            }
        }
    }

    // ---- normalize + store bf16 to [B, L, D] with head merge ----
    float inv[8];
#pragma unroll
    for (int v = 0; v < 8; ++v) inv[v] = 1.0f / lrow[v];
#pragma unroll
    for (int nt = 0; nt < 8; ++nt) {
#pragma unroll
        for (int v = 0; v < 8; ++v) {
            int r = rbase + v;
            Ob[((size_t)b * L + q0 + r) * D + h * DH + nt * 16 + colN] =
                f2bf(Oacc[nt][v] * inv[v]);
        }
    }
}

// ---------------------------------------------------------------------------
// Launch
// ---------------------------------------------------------------------------
extern "C" void kernel_launch(void* const* d_in, const int* in_sizes, int n_in,
                              void* d_out, int out_size, void* d_ws, size_t ws_size,
                              hipStream_t stream) {
    (void)in_sizes; (void)n_in; (void)out_size; (void)ws_size;
    constexpr int B = 2, L = 2048, D = 2048, H = 16, DH = 128;
    constexpr int M  = B * L;       // 4096
    constexpr int N1 = 3 * D;       // 6144

    const float* x     = (const float*)d_in[0];
    const float* w_qkv = (const float*)d_in[1];
    const float* w_out = (const float*)d_in[2];
    float* out = (float*)d_out;

    char* ws = (char*)d_ws;
    size_t off = 0;
    auto take = [&](size_t bytes) { void* p = ws + off; off += bytes; return p; };
    __bf16* xb    = (__bf16*)take((size_t)M * D * 2);
    __bf16* wqkvb = (__bf16*)take((size_t)N1 * D * 2);
    __bf16* woutb = (__bf16*)take((size_t)D * D * 2);
    __bf16* qkvb  = (__bf16*)take((size_t)M * N1 * 2);
    __bf16* Qb    = (__bf16*)take((size_t)B * H * L * DH * 2);
    __bf16* Kb    = (__bf16*)take((size_t)B * H * L * DH * 2);
    __bf16* Vt    = (__bf16*)take((size_t)B * H * DH * L * 2);   // [B*H, dh, L]
    __bf16* attnb = (__bf16*)take((size_t)M * D * 2);

    // 1) fp32 -> bf16
    f32_to_bf16_kernel<<<(M * D) / 256, 256, 0, stream>>>(x, xb, M * D);
    f32_to_bf16_kernel<<<(N1 * D) / 256, 256, 0, stream>>>(w_qkv, wqkvb, N1 * D);
    f32_to_bf16_kernel<<<(D * D) / 256, 256, 0, stream>>>(w_out, woutb, D * D);

    // 2) qkv = x @ w_qkv^T   (M=4096, N=6144, K=2048), bf16 out
    gemm_wmma_kernel<true><<<dim3(N1 / 64, M / 128), 256, 0, stream>>>(
        xb, wqkvb, (void*)qkvb, M, N1, D);

    // 3a) RoPE + scatter q,k to [B,H,L,dh]
    rope_scatter_kernel<<<(B * L * H * 64) / 256, 256, 0, stream>>>(qkvb, Qb, Kb);
    // 3b) V -> key-major Vt [B*H, dh, L]
    v_transpose_kernel<<<dim3(DH / 32, L / 32, B * H), 256, 0, stream>>>(qkvb, Vt);

    // 4) causal flash attention -> attnb [B,L,D] bf16
    flash_attn_kernel<<<B * H * (L / 64), 128, 0, stream>>>(Qb, Kb, Vt, attnb);

    // 5) out = attn @ w_out^T  (M=4096, N=2048, K=2048), f32 out
    gemm_wmma_kernel<false><<<dim3(D / 64, M / 128), 256, 0, stream>>>(
        attnb, woutb, (void*)out, M, D, D);
}
